// EquivariantCrossAttention_77790447665426
// MI455X (gfx1250) — matrix operand order repaired and checked by reading, not compiled, not run on hardware
//
#include <hip/hip_runtime.h>
#include <math.h>

// ---------------------------------------------------------------------------
// EquivariantCrossAttention, MI455X (gfx1250, wave32, WMMA).
//
// Algebraic reductions vs reference:
//  * q-projection folded into k:  Wk[l,h,e] = sum_d W_q[e,h*128+d] * k[l,h,d]
//    -> att logits = emb_q . Wk                      (saves ~17 GFLOP)
//  * W_mix pulled out of the latent sum (softmax sums to 1):
//    y = (sum_l att * v) @ W_mix + b_mix             (saves ~17 GFLOP)
//  * cond_b half factored out of the latent GEMM (linearity):
//    u_b = (att^T @ emb_v) @ W_ev_b + b_ev_b         (saves ~17 GFLOP)
// Remaining hot loop: emb_v(131072x128) @ W_ev_g(128x512) -> f16 WMMA,
// with the att*v0 weighted latent reduction done branchless on C-fragments.
// ---------------------------------------------------------------------------

typedef __attribute__((ext_vector_type(16))) _Float16 v16h;
typedef __attribute__((ext_vector_type(8)))  _Float16 v8h;
typedef __attribute__((ext_vector_type(8)))  float    v8f;

#define TWO_PI_F   6.283185307179586f
#define STDQ_F     2.0f
#define STDV_F     10.0f
#define ATT_SCALE  0.08838834764831845f   // 1/sqrt(128)
#define EMB_STRIDE 136                    // halves; 272B row stride, 16B aligned

__device__ __forceinline__ v8f wmma_f16(v16h a, v16h b, v8f c) {
    return __builtin_amdgcn_wmma_f32_16x16x32_f16(false, a, false, b,
                                                  (short)0, c, false, false);
}

// A-fragment load from LDS in contiguous k order (same order as convB B-frags):
// lane row M = lane%16; lanes 0-15 hold K {0..15}, lanes 16-31 {16..31}.
__device__ __forceinline__ v16h load_afrag(const _Float16* row, int kt, int hi16) {
    const _Float16* ap = row + kt * 32 + hi16 * 16;
    v8h lo = *(const v8h*)(ap);
    v8h hv = *(const v8h*)(ap + 8);
    return __builtin_shufflevector(lo, hv, 0,1,2,3,4,5,6,7,8,9,10,11,12,13,14,15);
}

// ---------------------------------------------------------------------------
// Precompute kernels
// ---------------------------------------------------------------------------

// k,v0 = c @ W_kv + b_kv   (L=128 rows x 1024 cols)
__global__ __launch_bounds__(256)
void kv_kernel(const float* __restrict__ c, const float* __restrict__ Wkv,
               const float* __restrict__ bkv,
               float* __restrict__ kd, float* __restrict__ v0)
{
    const int tid = blockIdx.x * 256 + threadIdx.x;   // 131072 total
    const int col = tid & 1023, l = tid >> 10;
    float sum = bkv[col];
    #pragma unroll 8
    for (int e = 0; e < 128; ++e) sum += c[l * 128 + e] * Wkv[e * 1024 + col];
    if (col < 512) kd[l * 512 + col] = sum;
    else           v0[l * 512 + (col - 512)] = sum;
}

// Wk[l,h,e] = sum_d W_q[e, h*128+d] * k[l, h*128+d];  bk[l,h] = b_q . k
__global__ __launch_bounds__(256)
void wk_kernel(const float* __restrict__ Wq, const float* __restrict__ bq,
               const float* __restrict__ kd,
               float* __restrict__ Wk, float* __restrict__ bk)
{
    const int tid = blockIdx.x * 256 + threadIdx.x;   // 65536 total
    const int e = tid & 127, lh = tid >> 7;
    const int h = lh & 3;
    const float* kp = kd + (lh >> 2) * 512 + h * 128;
    float sum = 0.f;
    #pragma unroll 8
    for (int d = 0; d < 128; ++d) sum += Wq[e * 512 + h * 128 + d] * kp[d];
    Wk[lh * 128 + e] = sum;
    if (e == 0) {
        float bs = 0.f;
        for (int d = 0; d < 128; ++d) bs += bq[h * 128 + d] * kp[d];
        bk[lh] = bs;
    }
}

// Convert f32 row-major (K x ncols) weight into f16 WMMA B-fragments:
// frag f = col_tile*KT + kt; inside: lane*16 + hi halves, where
// K = kt*32 + (lane/16)*16 + hi, col = col_tile*16 + lane%16.
__global__ __launch_bounds__(256)
void convB_kernel(const float* __restrict__ src, _Float16* __restrict__ dst,
                  int ncols, int KT, int total)
{
    const int tid = blockIdx.x * 256 + threadIdx.x;
    if (tid >= total) return;
    const int hi   = tid & 15;
    const int lane = (tid >> 4) & 31;
    const int f    = tid >> 9;
    const int kt   = f % KT, gct = f / KT;
    const int k    = kt * 32 + (lane >> 4) * 16 + hi;
    const int col  = gct * 16 + (lane & 15);
    dst[tid] = (_Float16)src[(size_t)k * ncols + col];
}

// Pre-swizzle v0 (128x512 f32) into WMMA C-fragment layout so the weighted
// latent reduction reads one 32B vector per lane per tile:
// tile = mt*32 + gct; element: row l = mt*16 + (lane/16)*8 + i, col = gct*16 + lane%16.
__global__ __launch_bounds__(256)
void v0c_kernel(const float* __restrict__ v0, float* __restrict__ v0C)
{
    const int tid = blockIdx.x * 256 + threadIdx.x;   // 65536 total
    const int i    = tid & 7;
    const int lane = (tid >> 3) & 31;
    const int tile = tid >> 8;
    const int mt = tile >> 5, gct = tile & 31;
    const int l = mt * 16 + (lane >> 4) * 8 + i;
    const int c = gct * 16 + (lane & 15);
    v0C[tid] = v0[l * 512 + c];
}

// ---------------------------------------------------------------------------
// Fused main kernel: one workgroup (8 wave32) per query point n.
// ---------------------------------------------------------------------------
__global__ __launch_bounds__(256)
void fused_attn_kernel(const float* __restrict__ x, const float* __restrict__ p,
                       const float* __restrict__ g,
                       const float* __restrict__ Wrffq, const float* __restrict__ Wrffv,
                       const float* __restrict__ bev, const float* __restrict__ Wev,
                       const float* __restrict__ Wk, const float* __restrict__ bk,
                       const _Float16* __restrict__ WevB, const float* __restrict__ v0C,
                       const float* __restrict__ Wmix, const float* __restrict__ bmix,
                       float* __restrict__ Y)
{
    __shared__ _Float16 embv[128 * EMB_STRIDE];   // (l=128) x (e=128) f16, padded
    __shared__ float    att_t[4 * 128];           // transposed: [h][l]
    __shared__ float    zred[512];                // z[h,e] = sum_l att*emb_v
    __shared__ float    u[1024];                  // [0..511]=g-part, [512..1023]=b-part

    const int n = blockIdx.x;
    const int t = threadIdx.x;
    const int l = t >> 1;
    const int s = t & 1;

    const float relx = x[n * 2 + 0] - p[l * 2 + 0];
    const float rely = x[n * 2 + 1] - p[l * 2 + 1];
    const float gl   = g[l];
    const float gw   = -(relx * relx + rely * rely) / (2.0f * gl * gl);

    // --- Phase A: attention logits (2 heads per thread) + emb_v into LDS ----
    {
        const int h0 = 2 * s, h1 = 2 * s + 1;
        const float* wk0 = Wk + (l * 4 + h0) * 128;
        const float* wk1 = Wk + (l * 4 + h1) * 128;
        float d0 = 0.f, d1 = 0.f;
        #pragma unroll 4
        for (int j = 0; j < 64; ++j) {
            const float hq = STDQ_F * TWO_PI_F * (relx * Wrffq[j] + rely * Wrffq[64 + j]);
            float sq, cq; __sincosf(hq, &sq, &cq);
            d0 += sq * wk0[j] + cq * wk0[64 + j];
            d1 += sq * wk1[j] + cq * wk1[64 + j];
        }
        att_t[h0 * 128 + l] = (d0 + bk[l * 4 + h0]) * ATT_SCALE + gw;
        att_t[h1 * 128 + l] = (d1 + bk[l * 4 + h1]) * ATT_SCALE + gw;

        const int j0 = s * 32;
        #pragma unroll 4
        for (int j = j0; j < j0 + 32; ++j) {
            const float hv = STDV_F * TWO_PI_F * (relx * Wrffv[j] + rely * Wrffv[64 + j]);
            float sv, cv; __sincosf(hv, &sv, &cv);
            embv[l * EMB_STRIDE + j]      = (_Float16)sv;
            embv[l * EMB_STRIDE + 64 + j] = (_Float16)cv;
        }
    }
    __syncthreads();

    // --- Phase B: softmax over latents, one wave32 per head ----------------
    {
        const int w = t >> 5, lane = t & 31;
        if (w < 4) {
            float v[4]; float m = -1e30f;
            #pragma unroll
            for (int r = 0; r < 4; ++r) { v[r] = att_t[w * 128 + lane + 32 * r]; m = fmaxf(m, v[r]); }
            #pragma unroll
            for (int off = 16; off > 0; off >>= 1) m = fmaxf(m, __shfl_xor(m, off));
            float ssum = 0.f;
            #pragma unroll
            for (int r = 0; r < 4; ++r) { v[r] = __expf(v[r] - m); ssum += v[r]; }
            #pragma unroll
            for (int off = 16; off > 0; off >>= 1) ssum += __shfl_xor(ssum, off);
            const float inv = 1.0f / ssum;
            #pragma unroll
            for (int r = 0; r < 4; ++r) att_t[w * 128 + lane + 32 * r] = v[r] * inv;
        }
    }
    __syncthreads();

    // --- Phase C (g-half): WMMA cond_g GEMM + branchless weighted reduction.
    // 8 waves share the 512 g-columns (4 col-tiles each).  Per C-tile the
    // weights att[l,h]*v0[l,c] are read as 32B fragments (LDS v8f + global v8f)
    // matching the accumulator layout: pure FMAs, no divergence.
    {
        const int w = t >> 5, lane = t & 31;
        const int col16 = lane & 15, hi16 = lane >> 4;
        for (int ct = 0; ct < 4; ++ct) {
            const int c_abs = w * 64 + ct * 16 + col16;
            const int gct   = w * 4 + ct;            // global col tile 0..31
            v16h bf[4];
            #pragma unroll
            for (int kt = 0; kt < 4; ++kt)
                bf[kt] = *(const v16h*)(WevB + (size_t)(gct * 4 + kt) * 512 + lane * 16);
            const float cadd = 1.0f + bev[c_abs];    // 1 + b_ev_g[c]
            const int   hsel = c_abs >> 7;           // tile-uniform head
            float colsum = 0.f;
            for (int mt = 0; mt < 8; ++mt) {         // latent tiles (GEMM M dim)
                v8f acc = {};
                const _Float16* arow = &embv[(mt * 16 + col16) * EMB_STRIDE];
                #pragma unroll
                for (int kt = 0; kt < 4; ++kt)
                    acc = wmma_f16(load_afrag(arow, kt, hi16), bf[kt], acc);
                const int rowbase = mt * 16 + hi16 * 8;   // C layout: VGPR i -> M=i(+8)
                const v8f attf = *(const v8f*)(&att_t[hsel * 128 + rowbase]);
                const v8f v0f  = *(const v8f*)(v0C + (size_t)((mt * 32 + gct) * 32 + lane) * 8);
                #pragma unroll
                for (int i = 0; i < 8; ++i)
                    colsum += (attf[i] * v0f[i]) * (acc[i] + cadd);
            }
            colsum += __shfl_xor(colsum, 16);        // combine the two row halves
            if (lane < 16) u[c_abs] = colsum;
        }
    }

    // --- Phase C2: z[h,e] = sum_l att[l,h] * emb_v[l,e]  (b-half factored) --
    #pragma unroll
    for (int rep = 0; rep < 2; ++rep) {
        const int o = t + rep * 256;
        const int h = o >> 7, e = o & 127;
        float zs = 0.f;
        #pragma unroll 8
        for (int ll = 0; ll < 128; ++ll)
            zs += att_t[h * 128 + ll] * (float)embv[ll * EMB_STRIDE + e];
        zred[o] = zs;
    }
    __syncthreads();

    // --- Phase C3: u_b[hd] = z[h,:] @ W_ev[:, 512+hd] + b_ev[512+hd] --------
    #pragma unroll
    for (int rep = 0; rep < 2; ++rep) {
        const int hd = t + rep * 256;
        const int h = hd >> 7;
        float s2 = bev[512 + hd];
        #pragma unroll 8
        for (int e = 0; e < 128; ++e)
            s2 += zred[h * 128 + e] * Wev[e * 1024 + 512 + hd];
        u[512 + hd] = s2;
    }
    __syncthreads();

    // --- Phase D: y = (u_g + u_b) @ W_mix + b_mix (per head), to workspace --
    #pragma unroll
    for (int rep = 0; rep < 2; ++rep) {
        const int cp = t + rep * 256;
        const int h = cp >> 7, dp = cp & 127;
        const float* ug = &u[h * 128];
        const float* ub = &u[512 + h * 128];
        float sum = bmix[dp];
        #pragma unroll 8
        for (int d = 0; d < 128; ++d) sum += (ug[d] + ub[d]) * Wmix[d * 128 + dp];
        Y[(size_t)n * 512 + cp] = sum;
    }
}

// ---------------------------------------------------------------------------
// Output projection: out(1024x512) = Y(1024x512) @ W_out(512x512) + b_out
// WMMA GEMM, 128x128 tile per block, K staged through LDS in 128-chunks.
// ---------------------------------------------------------------------------
__global__ __launch_bounds__(256)
void out_gemm_kernel(const float* __restrict__ Y, const _Float16* __restrict__ WoutB,
                     const float* __restrict__ bout, float* __restrict__ out)
{
    __shared__ _Float16 aY[128 * EMB_STRIDE];
    const int bm = blockIdx.x >> 2;          // 8 row blocks
    const int bn = blockIdx.x & 3;           // 4 col blocks
    const int t = threadIdx.x;
    const int w = t >> 5, lane = t & 31;
    const int col16 = lane & 15, hi16 = lane >> 4;

    v8f acc[8];
    #pragma unroll
    for (int ct = 0; ct < 8; ++ct) { v8f z = {}; acc[ct] = z; }

    for (int kc = 0; kc < 4; ++kc) {         // K chunks of 128
        {
            const int r = t >> 1, half = t & 1;
            const float* src = Y + (size_t)(bm * 128 + r) * 512 + kc * 128 + half * 64;
            _Float16* dst = &aY[r * EMB_STRIDE + half * 64];
            #pragma unroll 8
            for (int j = 0; j < 64; ++j) dst[j] = (_Float16)src[j];
        }
        __syncthreads();
        #pragma unroll
        for (int kt = 0; kt < 4; ++kt) {
            const _Float16* arow = &aY[(w * 16 + col16) * EMB_STRIDE];
            const v16h a = load_afrag(arow, kt, hi16);
            const int ktg = kc * 4 + kt;
            #pragma unroll
            for (int ct = 0; ct < 8; ++ct) {
                const int gct = bn * 8 + ct;
                const v16h b = *(const v16h*)(WoutB + (size_t)(gct * 16 + ktg) * 512 + lane * 16);
                acc[ct] = wmma_f16(a, b, acc[ct]);
            }
        }
        __syncthreads();
    }
    #pragma unroll
    for (int ct = 0; ct < 8; ++ct) {
        const int col = bn * 128 + ct * 16 + col16;
        const float bo = bout[col];
        const int rowbase = bm * 128 + w * 16 + hi16 * 8;
        #pragma unroll
        for (int i = 0; i < 8; ++i)
            out[(size_t)(rowbase + i) * 512 + col] = acc[ct][i] + bo;
    }
}

// ---------------------------------------------------------------------------
extern "C" void kernel_launch(void* const* d_in, const int* in_sizes, int n_in,
                              void* d_out, int out_size, void* d_ws, size_t ws_size,
                              hipStream_t stream)
{
    const float* x     = (const float*)d_in[0];
    const float* p     = (const float*)d_in[1];
    const float* c     = (const float*)d_in[2];
    const float* g     = (const float*)d_in[3];
    const float* Wrffq = (const float*)d_in[4];
    const float* Wrffv = (const float*)d_in[5];
    const float* Wq    = (const float*)d_in[6];
    const float* bq    = (const float*)d_in[7];
    const float* Wkv   = (const float*)d_in[8];
    const float* bkv   = (const float*)d_in[9];
    const float* Wev   = (const float*)d_in[10];
    const float* bev   = (const float*)d_in[11];
    const float* Wmix  = (const float*)d_in[12];
    const float* bmix  = (const float*)d_in[13];
    const float* Wout  = (const float*)d_in[14];
    const float* bout  = (const float*)d_in[15];

    char* ws = (char*)d_ws;
    float*    kd    = (float*)(ws + 0);          // 128x512 f32
    float*    v0    = (float*)(ws + 262144);     // 128x512 f32
    float*    Wk    = (float*)(ws + 524288);     // 128x4x128 f32
    float*    bk    = (float*)(ws + 786432);     // 512 f32
    float*    Y     = (float*)(ws + 788480);     // 1024x512 f32
    _Float16* WevB  = (_Float16*)(ws + 2885632); // 128x512 (g-half) f16 B-frags
    _Float16* WoutB = (_Float16*)(ws + 3016704); // 512x512 f16 B-frags
    float*    v0C   = (float*)(ws + 3540992);    // 128x512 f32, C-fragment layout

    kv_kernel<<<512, 256, 0, stream>>>(c, Wkv, bkv, kd, v0);
    wk_kernel<<<256, 256, 0, stream>>>(Wq, bq, kd, Wk, bk);
    v0c_kernel<<<256, 256, 0, stream>>>(v0, v0C);
    convB_kernel<<<256, 256, 0, stream>>>(Wev, WevB, 1024, 4, 65536);   // g-half only
    convB_kernel<<<1024, 256, 0, stream>>>(Wout, WoutB, 512, 16, 262144);
    fused_attn_kernel<<<1024, 256, 0, stream>>>(x, p, g, Wrffq, Wrffv, bev, Wev,
                                                Wk, bk, WevB, v0C, Wmix, bmix, Y);
    out_gemm_kernel<<<32, 256, 0, stream>>>(Y, WoutB, bout, (float*)d_out);
}
